// PerformerAttention_77429670412401
// MI455X (gfx1250) — matrix operand order repaired and checked by reading, not compiled, not run on hardware
//
#include <hip/hip_runtime.h>

#define DIM    1024
#define HEADS  16
#define HD     64
#define SEQ    4096
#define BATCH  4
#define MTOT   (BATCH * SEQ)   // 16384
#define KSPLIT 8
#define NCHUNK (SEQ / KSPLIT)  // 512

typedef __bf16 v16bf __attribute__((ext_vector_type(16)));
typedef float  v8f   __attribute__((ext_vector_type(8)));

union FragU { unsigned u[8]; v16bf v; };

#if __has_builtin(__builtin_amdgcn_global_load_async_to_lds_b128)
#define HAS_ASYNC_LDS 1
typedef int v4i_t __attribute__((ext_vector_type(4)));
typedef __attribute__((address_space(1))) v4i_t as1_v4i;   // global int4
typedef __attribute__((address_space(3))) v4i_t as3_v4i;   // LDS int4
#endif

__device__ __forceinline__ unsigned short f2bf(float f) {
  return __builtin_bit_cast(unsigned short, (__bf16)f);   // RNE, v_cvt capable
}
__device__ __forceinline__ float bf2f(unsigned short h) {
  return __builtin_bit_cast(float, (unsigned)h << 16);
}
__device__ __forceinline__ unsigned pack_bf2(float lo, float hi) {
  return (unsigned)f2bf(lo) | ((unsigned)f2bf(hi) << 16);
}
__device__ __forceinline__ unsigned long long pack_bf4(float4 v) {
  return (unsigned long long)pack_bf2(v.x, v.y) |
         ((unsigned long long)pack_bf2(v.z, v.w) << 32);
}

// A fragment from a 16x32 bf16 LDS tile, row-major [m][k] (row stride 32).
// ISA 7.12.2 16-bit A 16x32: lanes 0-15 K=0..7,16..23; lanes 16-31 K=8..15,24..31.
__device__ __forceinline__ v16bf load_fragA(const unsigned short* tile, int lane) {
  FragU f;
  const unsigned* row = (const unsigned*)(tile + (lane & 15) * 32);
  const int g4 = (lane >> 4) * 4;
#pragma unroll
  for (int r = 0; r < 4; ++r) {
    f.u[r]     = row[r + g4];
    f.u[4 + r] = row[r + 8 + g4];
  }
  return f.v;
}

// B fragment: tile holds B^T row-major [n][k] (row stride 32).
// lanes 0-15 hold K=0..7,16..23, lanes 16-31 hold K=8..15,24..31; N = lane%16.
__device__ __forceinline__ v16bf load_fragB(const unsigned short* tile, int lane) {
  FragU f;
  const unsigned* row = (const unsigned*)(tile + (lane & 15) * 32);
  const int g4 = (lane >> 4) * 4;
#pragma unroll
  for (int r = 0; r < 4; ++r) {
    f.u[r]     = row[r + g4];
    f.u[4 + r] = row[r + 8 + g4];
  }
  return f.v;
}

#define WMMA_BF16(A, B, C) \
  __builtin_amdgcn_wmma_f32_16x16x32_bf16(false, (A), false, (B), (short)0, (C), false, false)

// ---------------- Kernel 1: QKV projection + Performer feature map ----------
// Block tile 128(M) x 64(N), 128 threads / 4 waves; wave tile 32x64 (8 wmma/step).
// Writes q [bh][n][d], kT [bh][d][n], vT [bh][d][n] (transposed for KtV GEMM).
__global__ __launch_bounds__(128) void qkv_phi_kernel(
    const float* __restrict__ x, const float* __restrict__ Wqkv,
    unsigned short* __restrict__ qbuf, unsigned short* __restrict__ kT,
    unsigned short* __restrict__ vT) {
  __shared__ unsigned short ldsA[128 * 32];
  __shared__ unsigned short ldsB[64 * 32];
  const int tid = threadIdx.x, lane = tid & 31, wave = tid >> 5;
  const int mTile = blockIdx.y * 128, nTile = blockIdx.x * 64;

  v8f acc[2][4] = {};
  for (int kk = 0; kk < DIM; kk += 32) {
    __syncthreads();
#pragma unroll
    for (int t = 0; t < 8; ++t) {   // A: 128x32 = 1024 float4
      int idx = tid + t * 128;
      int r = idx >> 3, c4 = idx & 7;
      float4 v = *(const float4*)(x + (size_t)(mTile + r) * DIM + kk + c4 * 4);
      ((unsigned long long*)ldsA)[idx] = pack_bf4(v);
    }
#pragma unroll
    for (int t = 0; t < 4; ++t) {   // B: 64x32 = 512 float4
      int idx = tid + t * 128;
      int r = idx >> 3, c4 = idx & 7;
      float4 v = *(const float4*)(Wqkv + (size_t)(nTile + r) * DIM + kk + c4 * 4);
      ((unsigned long long*)ldsB)[idx] = pack_bf4(v);
    }
    __syncthreads();
    v16bf a0 = load_fragA(ldsA + (wave * 32) * 32, lane);
    v16bf a1 = load_fragA(ldsA + (wave * 32 + 16) * 32, lane);
#pragma unroll
    for (int j = 0; j < 4; ++j) {
      v16bf b = load_fragB(ldsB + j * 16 * 32, lane);
      acc[0][j] = WMMA_BF16(a0, b, acc[0][j]);
      acc[1][j] = WMMA_BF16(a1, b, acc[1][j]);
    }
  }
  const int nLane = lane & 15;
#pragma unroll
  for (int i = 0; i < 2; ++i) {
    const int mBase = mTile + wave * 32 + i * 16 + (lane >> 4) * 8;
#pragma unroll
    for (int j = 0; j < 4; ++j) {
      const int cG = nTile + j * 16 + nLane;
      const int part = cG >> 10, within = cG & 1023;
      const int head = within >> 6, dloc = within & 63;
#pragma unroll
      for (int r = 0; r < 8; ++r) {
        const int mG = mBase + r;
        const int bh = (mG >> 12) * HEADS + head;
        const int nn = mG & (SEQ - 1);
        float v = acc[i][j][r];
        if (part == 0) {
          v = __expf(-0.5f * v * v);
          qbuf[((size_t)bh * SEQ + nn) * HD + dloc] = f2bf(v);
        } else if (part == 1) {
          v = __expf(-0.5f * v * v);
          kT[((size_t)bh * HD + dloc) * SEQ + nn] = f2bf(v);
        } else {
          vT[((size_t)bh * HD + dloc) * SEQ + nn] = f2bf(v);
        }
      }
    }
  }
}

// ---------------- Kernel 2: split-K partial kv = K^T V, partial k_sum -------
// grid (64 bh, KSPLIT). Deterministic partial buffers. Staging uses CDNA5
// async global->LDS copies when the builtin is available.
__global__ __launch_bounds__(128) void kv_partial_kernel(
    const unsigned short* __restrict__ kT, const unsigned short* __restrict__ vT,
    float* __restrict__ kvPart, float* __restrict__ ksumPart) {
  __shared__ unsigned short ldsA[64 * 32];
  __shared__ unsigned short ldsB[64 * 32];
  const int bh = blockIdx.x, s = blockIdx.y;
  const int tid = threadIdx.x, lane = tid & 31, wave = tid >> 5;
  const unsigned short* kTp = kT + (size_t)bh * HD * SEQ;
  const unsigned short* vTp = vT + (size_t)bh * HD * SEQ;
  const int n0 = s * NCHUNK;

  v8f acc[4] = {};
  float ks = 0.f;
  for (int nn = n0; nn < n0 + NCHUNK; nn += 32) {
    __syncthreads();
#if HAS_ASYNC_LDS
#pragma unroll
    for (int t = 0; t < 2; ++t) {          // 256 16B chunks per tile
      int idx = tid + t * 128;
      int r = idx >> 2, c8 = idx & 3;      // 4 chunks per 64B row
      __builtin_amdgcn_global_load_async_to_lds_b128(
          (as1_v4i*)(kTp + (size_t)r * SEQ + nn + c8 * 8),
          (as3_v4i*)(ldsA + idx * 8), 0, 0);
      __builtin_amdgcn_global_load_async_to_lds_b128(
          (as1_v4i*)(vTp + (size_t)r * SEQ + nn + c8 * 8),
          (as3_v4i*)(ldsB + idx * 8), 0, 0);
    }
    asm volatile("s_wait_asynccnt 0x0" ::: "memory");
#else
#pragma unroll
    for (int t = 0; t < 4; ++t) {          // 512 u64 per tile
      int idx = tid + t * 128;
      int r = idx >> 3, c4 = idx & 7;
      ((unsigned long long*)ldsA)[idx] =
          *(const unsigned long long*)(kTp + (size_t)r * SEQ + nn + c4 * 4);
      ((unsigned long long*)ldsB)[idx] =
          *(const unsigned long long*)(vTp + (size_t)r * SEQ + nn + c4 * 4);
    }
#endif
    __syncthreads();
    if (tid < HD) {
#pragma unroll
      for (int c = 0; c < 32; ++c) ks += bf2f(ldsA[tid * 32 + c]);
    }
    v16bf a = load_fragA(ldsA + wave * 16 * 32, lane);
#pragma unroll
    for (int j = 0; j < 4; ++j) {
      v16bf b = load_fragB(ldsB + j * 16 * 32, lane);
      acc[j] = WMMA_BF16(a, b, acc[j]);
    }
  }
  float* kvDst = kvPart + ((size_t)s * (BATCH * HEADS) + bh) * HD * HD;
  const int dBase = wave * 16 + (lane >> 4) * 8;
  const int nLane = lane & 15;
#pragma unroll
  for (int j = 0; j < 4; ++j)
#pragma unroll
    for (int r = 0; r < 8; ++r)
      kvDst[(size_t)(dBase + r) * HD + j * 16 + nLane] = acc[j][r];
  if (tid < HD) ksumPart[((size_t)s * (BATCH * HEADS) + bh) * HD + tid] = ks;
}

// ---------------- Kernel 2b: reduce split-K partials ------------------------
__global__ __launch_bounds__(128) void kv_reduce_kernel(
    const float* __restrict__ kvPart, const float* __restrict__ ksumPart,
    float* __restrict__ kv, float* __restrict__ ksum) {
  const int bh = blockIdx.x, tid = threadIdx.x;
  for (int idx = tid; idx < HD * HD; idx += 128) {
    float s = 0.f;
#pragma unroll
    for (int p = 0; p < KSPLIT; ++p)
      s += kvPart[((size_t)p * (BATCH * HEADS) + bh) * HD * HD + idx];
    kv[(size_t)bh * HD * HD + idx] = s;
  }
  if (tid < HD) {
    float s = 0.f;
#pragma unroll
    for (int p = 0; p < KSPLIT; ++p)
      s += ksumPart[((size_t)p * (BATCH * HEADS) + bh) * HD + tid];
    ksum[(size_t)bh * HD + tid] = s;
  }
}

// ---------------- Kernel 3: out = (q @ kv) * z ------------------------------
// grid (SEQ/64, 64 bh). att written bf16 [B][n][C].
__global__ __launch_bounds__(128) void attn_out_kernel(
    const unsigned short* __restrict__ qbuf, const float* __restrict__ kv,
    const float* __restrict__ ksum, unsigned short* __restrict__ att) {
  __shared__ unsigned short ldsA[64 * 32];
  __shared__ unsigned short ldsB[64 * 32];
  __shared__ float ksumS[HD];
  __shared__ float zS[64];
  const int bh = blockIdx.y, rowTile = blockIdx.x * 64;
  const int bb = bh >> 4, head = bh & 15;
  const int tid = threadIdx.x, lane = tid & 31, wave = tid >> 5;
  const unsigned short* q = qbuf + (size_t)bh * SEQ * HD;
  const float* kvp = kv + (size_t)bh * HD * HD;

  if (tid < HD) ksumS[tid] = ksum[(size_t)bh * HD + tid];
  __syncthreads();
  if (tid < 64) {
    const unsigned short* qr = q + (size_t)(rowTile + tid) * HD;
    float sum = 0.f;
#pragma unroll
    for (int d = 0; d < HD; ++d) sum += bf2f(qr[d]) * ksumS[d];
    zS[tid] = 1.0f / (sum + 1e-6f);
  }
  v8f acc[4] = {};
#pragma unroll
  for (int kk = 0; kk < HD; kk += 32) {
    __syncthreads();
#pragma unroll
    for (int t = 0; t < 4; ++t) {          // A: 64x32 = 512 u64
      int idx = tid + t * 128;
      int r = idx >> 3, c4 = idx & 7;
      ((unsigned long long*)ldsA)[idx] =
          *(const unsigned long long*)(q + (size_t)(rowTile + r) * HD + kk + c4 * 4);
    }
#pragma unroll
    for (int t = 0; t < 8; ++t) {          // B^T = kv^T[e][d]: 1024 u32, packed pairs
      int idx = tid + t * 128;
      int e = idx >> 4, cp = idx & 15;
      ((unsigned*)ldsB)[idx] = pack_bf2(kvp[(size_t)(kk + 2 * cp) * HD + e],
                                        kvp[(size_t)(kk + 2 * cp + 1) * HD + e]);
    }
    __syncthreads();
    v16bf a = load_fragA(ldsA + wave * 16 * 32, lane);
#pragma unroll
    for (int j = 0; j < 4; ++j) {
      v16bf b = load_fragB(ldsB + j * 16 * 32, lane);
      acc[j] = WMMA_BF16(a, b, acc[j]);
    }
  }
  const int mBase = wave * 16 + (lane >> 4) * 8;
  const int nLane = lane & 15;
#pragma unroll
  for (int j = 0; j < 4; ++j)
#pragma unroll
    for (int r = 0; r < 8; ++r) {
      const int rloc = mBase + r;
      const int e = j * 16 + nLane;
      float v = acc[j][r] * zS[rloc];
      att[((size_t)bb * SEQ + rowTile + rloc) * DIM + head * HD + e] = f2bf(v);
    }
}

// ---------------- Kernel 4: output projection + bias ------------------------
// Block tile 128x64, wave tile 32x64 (8 wmma/step). grid (DIM/64, MTOT/128).
__global__ __launch_bounds__(128) void proj_kernel(
    const unsigned short* __restrict__ att, const float* __restrict__ Wp,
    const float* __restrict__ bias, float* __restrict__ out) {
  __shared__ unsigned short ldsA[128 * 32];
  __shared__ unsigned short ldsB[64 * 32];
  const int tid = threadIdx.x, lane = tid & 31, wave = tid >> 5;
  const int mTile = blockIdx.y * 128, nTile = blockIdx.x * 64;
  v8f acc[2][4] = {};
  for (int kk = 0; kk < DIM; kk += 32) {
    __syncthreads();
#pragma unroll
    for (int t = 0; t < 8; ++t) {          // A: 128x32 = 1024 u64 (bf16 copy)
      int idx = tid + t * 128;
      int r = idx >> 3, c4 = idx & 7;
      ((unsigned long long*)ldsA)[idx] =
          *(const unsigned long long*)(att + (size_t)(mTile + r) * DIM + kk + c4 * 4);
    }
#pragma unroll
    for (int t = 0; t < 4; ++t) {          // B: 64x32 = 512 float4 -> bf16
      int idx = tid + t * 128;
      int r = idx >> 3, c4 = idx & 7;
      float4 v = *(const float4*)(Wp + (size_t)(nTile + r) * DIM + kk + c4 * 4);
      ((unsigned long long*)ldsB)[idx] = pack_bf4(v);
    }
    __syncthreads();
    v16bf a0 = load_fragA(ldsA + (wave * 32) * 32, lane);
    v16bf a1 = load_fragA(ldsA + (wave * 32 + 16) * 32, lane);
#pragma unroll
    for (int j = 0; j < 4; ++j) {
      v16bf b = load_fragB(ldsB + j * 16 * 32, lane);
      acc[0][j] = WMMA_BF16(a0, b, acc[0][j]);
      acc[1][j] = WMMA_BF16(a1, b, acc[1][j]);
    }
  }
  const int nLane = lane & 15;
#pragma unroll
  for (int i = 0; i < 2; ++i) {
    const int mBase = mTile + wave * 32 + i * 16 + (lane >> 4) * 8;
#pragma unroll
    for (int j = 0; j < 4; ++j) {
      const int cG = nTile + j * 16 + nLane;
      const float bv = bias[cG];
#pragma unroll
      for (int r = 0; r < 8; ++r)
        out[(size_t)(mBase + r) * DIM + cG] = acc[i][j][r] + bv;
    }
  }
}

extern "C" void kernel_launch(void* const* d_in, const int* in_sizes, int n_in,
                              void* d_out, int out_size, void* d_ws, size_t ws_size,
                              hipStream_t stream) {
  (void)in_sizes; (void)n_in; (void)out_size; (void)ws_size;
  const float* x     = (const float*)d_in[0];
  const float* Wqkv  = (const float*)d_in[1];
  const float* Wproj = (const float*)d_in[2];
  const float* bproj = (const float*)d_in[3];
  float* out = (float*)d_out;

  char* ws = (char*)d_ws;
  size_t off = 0;
  const size_t headElems = (size_t)BATCH * HEADS * SEQ * HD;  // 16.7M
  unsigned short* qbuf = (unsigned short*)(ws + off); off += headElems * 2;          // 32 MB
  unsigned short* kT   = (unsigned short*)(ws + off); off += headElems * 2;          // 32 MB
  unsigned short* vT   = (unsigned short*)(ws + off); off += headElems * 2;          // 32 MB
  unsigned short* att  = (unsigned short*)(ws + off); off += (size_t)MTOT * DIM * 2; // 32 MB
  float* kvPart   = (float*)(ws + off); off += (size_t)KSPLIT * BATCH * HEADS * HD * HD * 4; // 8 MB
  float* ksumPart = (float*)(ws + off); off += (size_t)KSPLIT * BATCH * HEADS * HD * 4;
  float* kvbuf    = (float*)(ws + off); off += (size_t)BATCH * HEADS * HD * HD * 4;          // 1 MB
  float* ksumbuf  = (float*)(ws + off); off += (size_t)BATCH * HEADS * HD * 4;

  dim3 blk(128);
  qkv_phi_kernel<<<dim3(3 * DIM / 64, MTOT / 128), blk, 0, stream>>>(x, Wqkv, qbuf, kT, vT);
  kv_partial_kernel<<<dim3(BATCH * HEADS, KSPLIT), blk, 0, stream>>>(kT, vT, kvPart, ksumPart);
  kv_reduce_kernel<<<dim3(BATCH * HEADS), blk, 0, stream>>>(kvPart, ksumPart, kvbuf, ksumbuf);
  attn_out_kernel<<<dim3(SEQ / 64, BATCH * HEADS), blk, 0, stream>>>(qbuf, kvbuf, ksumbuf, att);
  proj_kernel<<<dim3(DIM / 64, MTOT / 128), blk, 0, stream>>>(att, Wproj, bproj, out);
}
